// GUEnc_38465727103472
// MI455X (gfx1250) — compile-verified, matrix operand order checked by reading, not run on hardware
//
#include <hip/hip_runtime.h>
#include <hip/hip_bf16.h>

typedef __attribute__((ext_vector_type(16))) _Float16 v16h;
typedef __attribute__((ext_vector_type(8)))  float    v8f;

#define CDIM 128
#define NHEADS 4
#define HD 32

// ---------------------------------------------------------------------------
// Weight pack: W[128x128] f32 (row-major, W[k*128+n]) -> f16 B-fragment layout.
// Fragment id = tile*4 + chunk (tile = n/16, chunk = k/32).
// Wp[(frag*32 + lane)*16 + t] = W[k*128 + n] with
//   n = tile*16 + (lane&15),  k = chunk*32 + (lane>=16?8:0) + (t<8 ? t : t+8)
// 16384 f16 total (32KB). Each lane's 16 halves are contiguous (32B).
// ---------------------------------------------------------------------------
__global__ void pack_w_kernel(const float* __restrict__ W, _Float16* __restrict__ Wp) {
  int idx = blockIdx.x * 256 + threadIdx.x;   // 0..16383
  int frag = idx >> 9;                        // 0..31
  int within = idx & 511;
  int lane = within >> 4;
  int t = within & 15;
  int tile = frag >> 2;
  int chunk = frag & 3;
  int lhalf = lane & 15;
  int k = chunk * 32 + (lane >= 16 ? 8 : 0) + (t < 8 ? t : t + 8);
  int n = tile * 16 + lhalf;
  Wp[idx] = (_Float16)W[k * CDIM + n];
}

// ---------------------------------------------------------------------------
// WMMA GEMM: Y[M x 128] = X[M x 128] @ W[128 x 128] (+bias)(+relu)
// grid.x = M/16, block = 128 threads (4 waves). Each wave owns 2 col tiles.
// X tile (16x128 f32, 8KB) staged to LDS via global_load_async_to_lds_b128;
// B operands pre-packed f16 fragments; f32 accumulate in v_wmma_f32_16x16x32_f16.
// ---------------------------------------------------------------------------
__global__ void wmma_gemm128(const float* __restrict__ X,
                             const _Float16* __restrict__ Wp,
                             const float* __restrict__ bias,
                             float* __restrict__ Y,
                             int M, int relu) {
  __shared__ float sx[16 * CDIM];

  const int row0  = blockIdx.x * 16;
  const int wave  = threadIdx.x >> 5;   // 0..3
  const int lane  = threadIdx.x & 31;
  const int lhalf = lane & 15;
  const bool hi   = lane >= 16;

  // --- async-stage the 16x128 f32 A tile into LDS (8KB = 512 x b128) ---
  {
    unsigned ldsb = (unsigned)(unsigned long long)(void*)&sx[0]; // LDS byte offset
    const float* xt = X + (long)row0 * CDIM;
#pragma unroll
    for (int p = 0; p < 4; ++p) {
      int off16 = p * 128 + threadIdx.x;  // 16-byte chunk index
      unsigned loff = ldsb + off16 * 16;
      unsigned long long g = (unsigned long long)(const void*)(xt + off16 * 4);
      asm volatile("global_load_async_to_lds_b128 %0, %1, off"
                   :: "v"(loff), "v"(g) : "memory");
    }
    asm volatile("s_wait_asynccnt 0x0" ::: "memory");
  }
  __syncthreads();

  v8f acc0 = {};
  v8f acc1 = {};
  // base of this wave's packed B fragments: tiles 2*wave and 2*wave+1
  const _Float16* wp0 = Wp + ((size_t)(2 * wave) * 4) * 512 + (size_t)lane * 16;

  for (int kc = 0; kc < 128; kc += 32) {
    // A fragment from LDS: row m=lhalf; lane<16 -> K={kc+0..7, kc+16..23}; hi -> +8
    v16h a;
    const float* ar = &sx[lhalf * CDIM + kc + (hi ? 8 : 0)];
#pragma unroll
    for (int t = 0; t < 8; ++t) {
      a[t]     = (_Float16)ar[t];
      a[t + 8] = (_Float16)ar[t + 16];
    }
    const int chunk = kc >> 5;
    v16h b0 = *(const v16h*)(wp0 + (size_t)chunk * 512);
    v16h b1 = *(const v16h*)(wp0 + (size_t)chunk * 512 + 2048);
    acc0 = __builtin_amdgcn_wmma_f32_16x16x32_f16(false, a, false, b0, (short)0, acc0, false, false);
    acc1 = __builtin_amdgcn_wmma_f32_16x16x32_f16(false, a, false, b1, (short)0, acc1, false, false);
  }

  // C/D layout: VGPR i -> row (hi?8:0)+i, lane -> column
  const int c0 = wave * 32 + lhalf;
  const int mbase = row0 + (hi ? 8 : 0);
#pragma unroll
  for (int i = 0; i < 8; ++i) {
    float v0 = acc0[i], v1 = acc1[i];
    if (bias) { v0 += bias[c0]; v1 += bias[c0 + 16]; }
    if (relu) { v0 = fmaxf(v0, 0.f); v1 = fmaxf(v1, 0.f); }
    Y[(long)(mbase + i) * CDIM + c0]      = v0;
    Y[(long)(mbase + i) * CDIM + c0 + 16] = v1;
  }
}

// ---------------------------------------------------------------------------
// Elementwise / utility kernels
// ---------------------------------------------------------------------------
__global__ void fill_kernel(float* p, float v, long n) {
  long i = (long)blockIdx.x * 256 + threadIdx.x;
  if (i < n) p[i] = v;
}
__global__ void copy_kernel(const float* a, float* b, long n) {
  long i = (long)blockIdx.x * 256 + threadIdx.x;
  if (i < n) b[i] = a[i];
}
__global__ void relu_kernel(float* p, long n) {
  long i = (long)blockIdx.x * 256 + threadIdx.x;
  if (i < n) p[i] = fmaxf(p[i], 0.f);
}

// ---------------------------------------------------------------------------
// GCN pieces: deg scatter, rsqrt, self-loop term, edge scatter
// ---------------------------------------------------------------------------
__global__ void deg_kernel(const int* __restrict__ col, const float* __restrict__ w,
                           float* deg, int E) {
  int e = blockIdx.x * 256 + threadIdx.x;
  if (e < E) {
    float we = w[e];
    if (we != 0.f) atomicAdd(&deg[col[e]], we);
  }
}
__global__ void dis_kernel(float* deg, float fill, int n) {
  int i = blockIdx.x * 256 + threadIdx.x;
  if (i < n) deg[i] = rsqrtf(deg[i] + fill);
}
// out = fill*dis^2*h + b   (dis already includes fill in the degree)
__global__ void gcn_self_kernel(const float* __restrict__ h, const float* __restrict__ dis,
                                const float* __restrict__ b, float* __restrict__ out,
                                float fill, int n) {
  long i = (long)blockIdx.x * 256 + threadIdx.x;
  if (i < (long)n * CDIM) {
    int r = (int)(i >> 7), c = (int)(i & 127);
    float d = dis[r];
    out[i] = fill * d * d * h[i] + b[c];
  }
}
// out[col] += dis[row]*w*dis[col]*h[row]   32 threads per edge, 4 ch each
__global__ void gcn_edge_kernel(const int* __restrict__ row, const int* __restrict__ col,
                                const float* __restrict__ w, const float* __restrict__ dis,
                                const float* __restrict__ h, float* __restrict__ out, int E) {
  int idx = blockIdx.x * 256 + threadIdx.x;
  int e = idx >> 5;
  if (e >= E) return;
  float we = w[e];
  if (we == 0.f) return;
  int r = row[e], cN = col[e];
  int c4 = idx & 31;
  const float4 hv = ((const float4*)(h + (long)r * CDIM))[c4];
  float coef = dis[r] * we * dis[cN];
  float* o = out + (long)cN * CDIM + c4 * 4;
  atomicAdd(o + 0, coef * hv.x);
  atomicAdd(o + 1, coef * hv.y);
  atomicAdd(o + 2, coef * hv.z);
  atomicAdd(o + 3, coef * hv.w);
}

// ---------------------------------------------------------------------------
// TopK pooling: score, key pack, bitonic sort, extraction, remap
// ---------------------------------------------------------------------------
__global__ void score_kernel(const float* __restrict__ x, const float* __restrict__ pw,
                             float* score, int n) {
  __shared__ float spw[CDIM];
  __shared__ float inv;
  int tid = threadIdx.x;
  if (tid < CDIM) spw[tid] = pw[tid];
  __syncthreads();
  if (tid == 0) {
    float s = 0.f;
    for (int d = 0; d < CDIM; ++d) s += spw[d] * spw[d];
    inv = rsqrtf(s);
  }
  __syncthreads();
  int i = blockIdx.x * 256 + tid;
  if (i < n) {
    const float* xr = x + (long)i * CDIM;
    float s = 0.f;
    for (int d = 0; d < CDIM; ++d) s += xr[d] * spw[d];
    score[i] = tanhf(s * inv);
  }
}
__global__ void key_kernel(const float* score, unsigned long long* keys, int n) {
  int i = blockIdx.x * 256 + threadIdx.x;
  if (i < n) {
    unsigned int b = __float_as_uint(score[i]);
    unsigned int u = (b & 0x80000000u) ? ~b : (b | 0x80000000u); // ascending map
    unsigned int ku = ~u;                                       // descending
    keys[i] = ((unsigned long long)ku << 32) | (unsigned int)i;
  }
}
__global__ void bitonic_kernel(unsigned long long* keys, int n, int k, int j) {
  int i = blockIdx.x * 256 + threadIdx.x;
  if (i >= n) return;
  int ixj = i ^ j;
  if (ixj > i) {
    bool up = ((i & k) == 0);
    unsigned long long a = keys[i], b = keys[ixj];
    if ((a > b) == up) { keys[i] = b; keys[ixj] = a; }
  }
}
__global__ void perm_kernel(const unsigned long long* keys, int* perm, int k) {
  int i = blockIdx.x * 256 + threadIdx.x;
  if (i < k) perm[i] = (int)(keys[i] & 0xffffffffu);
}
__global__ void pool_x_kernel(const float* __restrict__ x, const float* __restrict__ score,
                              const int* __restrict__ perm, float* xn, int k) {
  long i = (long)blockIdx.x * 256 + threadIdx.x;
  if (i < (long)k * CDIM) {
    int r = (int)(i >> 7), c = (int)(i & 127);
    int p = perm[r];
    xn[i] = x[(long)p * CDIM + c] * score[p];
  }
}
__global__ void cluster_init_kernel(int* cl, int n) {
  int i = blockIdx.x * 256 + threadIdx.x;
  if (i < n) cl[i] = -1;
}
__global__ void cluster_set_kernel(int* cl, const int* perm, int k) {
  int i = blockIdx.x * 256 + threadIdx.x;
  if (i < k) cl[perm[i]] = i;
}
__global__ void edge_remap_kernel(const int* __restrict__ row, const int* __restrict__ col,
                                  const float* __restrict__ w, const int* __restrict__ cl,
                                  int* nrow, int* ncol, float* nw, int E) {
  int e = blockIdx.x * 256 + threadIdx.x;
  if (e >= E) return;
  int cr = cl[row[e]], cc = cl[col[e]];
  float we = w[e];
  bool keep = (cr >= 0) && (cc >= 0) && (we > 0.f);
  nrow[e] = keep ? cr : 0;
  ncol[e] = keep ? cc : 1;
  nw[e]   = keep ? we : 0.f;
}
// unpool: dst[perm[r]] += src[r]  (perm rows unique -> no atomics)
__global__ void add_rows_kernel(float* dst, const int* __restrict__ perm,
                                const float* __restrict__ src, int k) {
  long i = (long)blockIdx.x * 256 + threadIdx.x;
  if (i < (long)k * CDIM) {
    int r = (int)(i >> 7), c = (int)(i & 127);
    dst[(long)perm[r] * CDIM + c] += src[i];
  }
}

// ---------------------------------------------------------------------------
// GMT attention: Out = Q + softmax(Qh·Kh^T / sqrt(128)) @ Vh per head.
// grid = (nq, NHEADS), block = 256, two-pass softmax streamed over keys.
// Bandwidth-bound (0.26 FLOP/byte) -> VALU path, data streams through L2.
// ---------------------------------------------------------------------------
__global__ void attention_kernel(const float* __restrict__ Q, const float* __restrict__ Km,
                                 const float* __restrict__ Vm, float* __restrict__ Out,
                                 int nk) {
  const int q = blockIdx.x, h = blockIdx.y, tid = threadIdx.x;
  const int d0 = h * HD;
  __shared__ float qv[HD];
  __shared__ float red[256];
  __shared__ float outv[HD];
  if (tid < HD) qv[tid] = Q[(long)q * CDIM + d0 + tid];
  __syncthreads();
  const float scale = 0.08838834764831845f; // 1/sqrt(128): PyG scales by sqrt(dim_V)

  float m = -3.0e38f;
  for (int k = tid; k < nk; k += 256) {
    const float* kr = Km + (long)k * CDIM + d0;
    float s = 0.f;
#pragma unroll
    for (int d = 0; d < HD; ++d) s += qv[d] * kr[d];
    m = fmaxf(m, s * scale);
  }
  red[tid] = m; __syncthreads();
  for (int o = 128; o > 0; o >>= 1) { if (tid < o) red[tid] = fmaxf(red[tid], red[tid + o]); __syncthreads(); }
  m = red[0]; __syncthreads();

  float l = 0.f;
  float acc[HD];
#pragma unroll
  for (int d = 0; d < HD; ++d) acc[d] = 0.f;
  for (int k = tid; k < nk; k += 256) {
    const float* kr = Km + (long)k * CDIM + d0;
    float s = 0.f;
#pragma unroll
    for (int d = 0; d < HD; ++d) s += qv[d] * kr[d];
    float e = __expf(s * scale - m);
    l += e;
    const float* vr = Vm + (long)k * CDIM + d0;
#pragma unroll
    for (int d = 0; d < HD; ++d) acc[d] += e * vr[d];
  }
  red[tid] = l; __syncthreads();
  for (int o = 128; o > 0; o >>= 1) { if (tid < o) red[tid] += red[tid + o]; __syncthreads(); }
  l = red[0]; __syncthreads();
  for (int d = 0; d < HD; ++d) {
    red[tid] = acc[d]; __syncthreads();
    for (int o = 128; o > 0; o >>= 1) { if (tid < o) red[tid] += red[tid + o]; __syncthreads(); }
    if (tid == 0) outv[d] = red[0];
    __syncthreads();
  }
  if (tid < HD)
    Out[(long)q * CDIM + d0 + tid] = Q[(long)q * CDIM + d0 + tid] + outv[tid] / l;
}

// ---------------------------------------------------------------------------
// Small GEMM: Y = (addsrc?addsrc:0) + act(X[rows x K] @ W[K x cols] + b)
// ---------------------------------------------------------------------------
__global__ void small_gemm_kernel(const float* __restrict__ X, const float* __restrict__ W,
                                  const float* __restrict__ b, const float* __restrict__ addsrc,
                                  float* Y, int rows, int K, int cols, int relu) {
  int idx = blockIdx.x * 256 + threadIdx.x;
  if (idx >= rows * cols) return;
  int r = idx / cols, c = idx % cols;
  float s = b ? b[c] : 0.f;
  for (int k = 0; k < K; ++k) s += X[(long)r * K + k] * W[(long)k * cols + c];
  if (relu) s = fmaxf(s, 0.f);
  if (addsrc) s += addsrc[(long)r * cols + c];
  Y[idx] = s;
}

// ---------------------------------------------------------------------------
// Host orchestration
// ---------------------------------------------------------------------------
extern "C" void kernel_launch(void* const* d_in, const int* in_sizes, int n_in,
                              void* d_out, int out_size, void* d_ws, size_t ws_size,
                              hipStream_t stream) {
  (void)in_sizes; (void)n_in; (void)out_size; (void)ws_size;
  const int N = 32768, E = 262144;
  const int KS[3] = {16384, 8192, 4096};

  const float* x      = (const float*)d_in[0];
  const int*   e0row  = (const int*)d_in[1];
  const int*   e0col  = e0row + E;
  const float* down_w[4] = {(const float*)d_in[2], (const float*)d_in[3], (const float*)d_in[4], (const float*)d_in[5]};
  const float* down_b[4] = {(const float*)d_in[6], (const float*)d_in[7], (const float*)d_in[8], (const float*)d_in[9]};
  const float* pool_w[3] = {(const float*)d_in[10], (const float*)d_in[11], (const float*)d_in[12]};
  const float* up_w[3]   = {(const float*)d_in[13], (const float*)d_in[14], (const float*)d_in[15]};
  const float* up_b[3]   = {(const float*)d_in[16], (const float*)d_in[17], (const float*)d_in[18]};
  const float* lin1_w = (const float*)d_in[19]; const float* lin1_b = (const float*)d_in[20];
  const float* S1     = (const float*)d_in[21]; const float* S3     = (const float*)d_in[22];
  const float* lin2_w = (const float*)d_in[23]; const float* lin2_b = (const float*)d_in[24];
  const float* out_w  = (const float*)d_in[25]; const float* out_b  = (const float*)d_in[26];
  const float* g1p[8]; const float* s2p[8]; const float* g3p[8];
  for (int i = 0; i < 8; ++i) { g1p[i] = (const float*)d_in[27 + i]; s2p[i] = (const float*)d_in[35 + i]; g3p[i] = (const float*)d_in[43 + i]; }

  char* base = (char*)d_ws;
  size_t off = 0;
  auto alloc = [&](size_t bytes) -> void* {
    void* p = base + off;
    off += (bytes + 255) & ~(size_t)255;
    return p;
  };
  float* ones_w = (float*)alloc((size_t)E * 4);
  _Float16* Wp = (_Float16*)alloc((size_t)16384 * 2);   // packed weight fragments
  float* hbuf = (float*)alloc((size_t)N * CDIM * 4);
  float* tmp  = (float*)alloc((size_t)N * CDIM * 4);
  float* bigA = (float*)alloc((size_t)N * CDIM * 4);
  float* bigB = (float*)alloc((size_t)N * CDIM * 4);
  float* xs0  = (float*)alloc((size_t)N * CDIM * 4);
  float* xs1  = (float*)alloc((size_t)KS[0] * CDIM * 4);
  float* xs2  = (float*)alloc((size_t)KS[1] * CDIM * 4);
  float* deg   = (float*)alloc((size_t)N * 4);
  float* score = (float*)alloc((size_t)N * 4);
  unsigned long long* keys = (unsigned long long*)alloc((size_t)N * 8);
  int* cluster = (int*)alloc((size_t)N * 4);
  int* perm0 = (int*)alloc((size_t)KS[0] * 4);
  int* perm1 = (int*)alloc((size_t)KS[1] * 4);
  int* perm2 = (int*)alloc((size_t)KS[2] * 4);
  int* erow[3]; int* ecol[3]; float* ew[3];
  for (int i = 0; i < 3; ++i) {
    erow[i] = (int*)alloc((size_t)E * 4);
    ecol[i] = (int*)alloc((size_t)E * 4);
    ew[i]   = (float*)alloc((size_t)E * 4);
  }
  float* q75 = (float*)alloc(80 * CDIM * 4);
  float* o75 = (float*)alloc(80 * CDIM * 4);
  float* kk75 = (float*)alloc(80 * CDIM * 4);
  float* vv75 = (float*)alloc(80 * CDIM * 4);
  float* zA  = (float*)alloc(80 * CDIM * 4);
  float* zB  = (float*)alloc(80 * CDIM * 4);
  float* q1  = (float*)alloc(CDIM * 4);
  float* o1  = (float*)alloc(CDIM * 4);
  float* z1  = (float*)alloc(CDIM * 4);
  float* z4  = (float*)alloc(CDIM * 4);

  auto g1d = [](long n) { return (unsigned)((n + 255) / 256); };

  fill_kernel<<<g1d(E), 256, 0, stream>>>(ones_w, 1.0f, E);

  auto run_gemm = [&](const float* xin, const float* W, const float* bias, float* out,
                      int n, int relu_) {
    pack_w_kernel<<<64, 256, 0, stream>>>(W, Wp);
    wmma_gemm128<<<dim3(n / 16), dim3(128), 0, stream>>>(xin, Wp, bias, out, n, relu_);
  };

  auto run_gcn = [&](const float* xin, const int* er, const int* ec, const float* w_,
                     const float* W, const float* B, float fill, int n, float* out, int relu_) {
    run_gemm(xin, W, nullptr, hbuf, n, 0);
    fill_kernel<<<g1d(n), 256, 0, stream>>>(deg, 0.f, n);
    deg_kernel<<<g1d(E), 256, 0, stream>>>(ec, w_, deg, E);
    dis_kernel<<<g1d(n), 256, 0, stream>>>(deg, fill, n);
    gcn_self_kernel<<<g1d((long)n * CDIM), 256, 0, stream>>>(hbuf, deg, B, out, fill, n);
    gcn_edge_kernel<<<g1d((long)E * 32), 256, 0, stream>>>(er, ec, w_, deg, hbuf, out, E);
    if (relu_) relu_kernel<<<g1d((long)n * CDIM), 256, 0, stream>>>(out, (long)n * CDIM);
  };

  auto run_topk = [&](const float* xin, const int* er, const int* ec, const float* w_,
                      const float* pw, int k, int n, float* xout, int* permbuf,
                      int* nr, int* nc, float* nw) {
    score_kernel<<<g1d(n), 256, 0, stream>>>(xin, pw, score, n);
    key_kernel<<<g1d(n), 256, 0, stream>>>(score, keys, n);
    for (int kk = 2; kk <= n; kk <<= 1)
      for (int j = kk >> 1; j > 0; j >>= 1)
        bitonic_kernel<<<g1d(n), 256, 0, stream>>>(keys, n, kk, j);
    perm_kernel<<<g1d(k), 256, 0, stream>>>(keys, permbuf, k);
    pool_x_kernel<<<g1d((long)k * CDIM), 256, 0, stream>>>(xin, score, permbuf, xout, k);
    cluster_init_kernel<<<g1d(n), 256, 0, stream>>>(cluster, n);
    cluster_set_kernel<<<g1d(k), 256, 0, stream>>>(cluster, permbuf, k);
    edge_remap_kernel<<<g1d(E), 256, 0, stream>>>(er, ec, w_, cluster, nr, nc, nw, E);
  };

  // ---- down path ----
  run_gcn(x, e0row, e0col, ones_w, down_w[0], down_b[0], 2.0f, N, xs0, 1);
  run_topk(xs0, e0row, e0col, ones_w, pool_w[0], KS[0], N, tmp, perm0, erow[0], ecol[0], ew[0]);
  run_gcn(tmp, erow[0], ecol[0], ew[0], down_w[1], down_b[1], 2.0f, KS[0], xs1, 1);
  run_topk(xs1, erow[0], ecol[0], ew[0], pool_w[1], KS[1], KS[0], tmp, perm1, erow[1], ecol[1], ew[1]);
  run_gcn(tmp, erow[1], ecol[1], ew[1], down_w[2], down_b[2], 2.0f, KS[1], xs2, 1);
  run_topk(xs2, erow[1], ecol[1], ew[1], pool_w[2], KS[2], KS[1], tmp, perm2, erow[2], ecol[2], ew[2]);
  run_gcn(tmp, erow[2], ecol[2], ew[2], down_w[3], down_b[3], 2.0f, KS[2], bigA, 1);

  // ---- up path (sum skip + unpool scatter) ----
  copy_kernel<<<g1d((long)KS[1] * CDIM), 256, 0, stream>>>(xs2, tmp, (long)KS[1] * CDIM);
  add_rows_kernel<<<g1d((long)KS[2] * CDIM), 256, 0, stream>>>(tmp, perm2, bigA, KS[2]);
  run_gcn(tmp, erow[1], ecol[1], ew[1], up_w[0], up_b[0], 2.0f, KS[1], bigB, 1);

  copy_kernel<<<g1d((long)KS[0] * CDIM), 256, 0, stream>>>(xs1, tmp, (long)KS[0] * CDIM);
  add_rows_kernel<<<g1d((long)KS[1] * CDIM), 256, 0, stream>>>(tmp, perm1, bigB, KS[1]);
  run_gcn(tmp, erow[0], ecol[0], ew[0], up_w[1], up_b[1], 2.0f, KS[0], bigA, 1);

  copy_kernel<<<g1d((long)N * CDIM), 256, 0, stream>>>(xs0, tmp, (long)N * CDIM);
  add_rows_kernel<<<g1d((long)KS[0] * CDIM), 256, 0, stream>>>(tmp, perm0, bigA, KS[0]);
  run_gcn(tmp, e0row, e0col, ones_w, up_w[2], up_b[2], 2.0f, N, bigB, 0);

  // ---- GraphMultisetTransformer readout ----
  run_gemm(bigB, lin1_w, lin1_b, bigA, N, 0);                              // h
  run_gcn(bigA, e0row, e0col, ones_w, g1p[2], g1p[3], 1.0f, N, xs0, 0);    // K1
  run_gcn(bigA, e0row, e0col, ones_w, g1p[4], g1p[5], 1.0f, N, tmp, 0);    // V1

  // GMPool_G
  small_gemm_kernel<<<g1d(75 * CDIM), 256, 0, stream>>>(S1, g1p[0], g1p[1], nullptr, q75, 75, CDIM, CDIM, 0);
  attention_kernel<<<dim3(75, NHEADS), 256, 0, stream>>>(q75, xs0, tmp, o75, N);
  small_gemm_kernel<<<g1d(75 * CDIM), 256, 0, stream>>>(o75, g1p[6], g1p[7], o75, zA, 75, CDIM, CDIM, 1);

  // SelfAtt
  small_gemm_kernel<<<g1d(75 * CDIM), 256, 0, stream>>>(zA, s2p[0], s2p[1], nullptr, q75, 75, CDIM, CDIM, 0);
  small_gemm_kernel<<<g1d(75 * CDIM), 256, 0, stream>>>(zA, s2p[2], s2p[3], nullptr, kk75, 75, CDIM, CDIM, 0);
  small_gemm_kernel<<<g1d(75 * CDIM), 256, 0, stream>>>(zA, s2p[4], s2p[5], nullptr, vv75, 75, CDIM, CDIM, 0);
  attention_kernel<<<dim3(75, NHEADS), 256, 0, stream>>>(q75, kk75, vv75, o75, 75);
  small_gemm_kernel<<<g1d(75 * CDIM), 256, 0, stream>>>(o75, s2p[6], s2p[7], o75, zB, 75, CDIM, CDIM, 1);

  // GMPool_I
  small_gemm_kernel<<<1, 256, 0, stream>>>(S3, g3p[0], g3p[1], nullptr, q1, 1, CDIM, CDIM, 0);
  small_gemm_kernel<<<g1d(75 * CDIM), 256, 0, stream>>>(zB, g3p[2], g3p[3], nullptr, kk75, 75, CDIM, CDIM, 0);
  small_gemm_kernel<<<g1d(75 * CDIM), 256, 0, stream>>>(zB, g3p[4], g3p[5], nullptr, vv75, 75, CDIM, CDIM, 0);
  attention_kernel<<<dim3(1, NHEADS), 256, 0, stream>>>(q1, kk75, vv75, o1, 75);
  small_gemm_kernel<<<1, 256, 0, stream>>>(o1, g3p[6], g3p[7], o1, z1, 1, CDIM, CDIM, 1);

  // head
  small_gemm_kernel<<<1, 256, 0, stream>>>(z1, lin2_w, lin2_b, nullptr, z4, 1, CDIM, CDIM, 0);
  small_gemm_kernel<<<1, 256, 0, stream>>>(z4, out_w, out_b, nullptr, (float*)d_out, 1, CDIM, 10, 0);
}
